// SieveGNN_25735444038221
// MI455X (gfx1250) — matrix operand order repaired and checked by reading, not compiled, not run on hardware
//
#include <hip/hip_runtime.h>
#include <hip/hip_bf16.h>
#include <math.h>

// ---------------------------------------------------------------------------
// 2-layer GAT for gfx1250 (MI455X).
//  - GEMMs via v_wmma_f32_16x16x32_bf16 (wave32, 16x16 C tile / wave).
//  - Segment softmax via encoded-float atomicMax + atomicAdd f32.
//  - Edge aggregation: float4 gathers of h[src] (L2-resident) + f32 atomics.
// ---------------------------------------------------------------------------

typedef __bf16 bf16;
typedef __attribute__((ext_vector_type(16))) __bf16 v16bf;
typedef __attribute__((ext_vector_type(8)))  float  v8f;

union Frag16 { uint4 u4[2]; v16bf v; };   // 32 bytes: 16 bf16 = 8 VGPRs/lane

__device__ __forceinline__ bf16 f2bf(float f) {
  // round-to-nearest-even f32 -> bf16
  unsigned u = __float_as_uint(f);
  unsigned r = u + 0x7FFFu + ((u >> 16) & 1u);
  unsigned short h = (unsigned short)(r >> 16);
  bf16 b;
  __builtin_memcpy(&b, &h, sizeof(b));
  return b;
}

// order-preserving float <-> u32 encoding for atomicMax
__device__ __forceinline__ unsigned fenc(float f) {
  unsigned u = __float_as_uint(f);
  return (u & 0x80000000u) ? ~u : (u | 0x80000000u);
}
__device__ __forceinline__ float fdec(unsigned u) {
  u = (u & 0x80000000u) ? (u & 0x7FFFFFFFu) : ~u;
  return __uint_as_float(u);
}

// --------------------------- elementwise helpers ---------------------------

__global__ void k_f32_to_bf16(const float* __restrict__ in, bf16* __restrict__ out,
                              unsigned n) {
  unsigned i = blockIdx.x * blockDim.x + threadIdx.x;
  if (i < n) out[i] = f2bf(in[i]);
}

__global__ void k_elu_to_bf16(const float* __restrict__ in, bf16* __restrict__ out,
                              unsigned n) {
  unsigned i = blockIdx.x * blockDim.x + threadIdx.x;
  if (i < n) {
    float v = in[i];
    v = (v > 0.f) ? v : (expf(v) - 1.f);
    out[i] = f2bf(v);
  }
}

__global__ void k_fill_u32(unsigned* __restrict__ p, unsigned v, unsigned n) {
  unsigned i = blockIdx.x * blockDim.x + threadIdx.x;
  if (i < n) p[i] = v;
}

__global__ void k_bias_init(float* __restrict__ out, const float* __restrict__ b,
                            unsigned n, int C) {
  unsigned i = blockIdx.x * blockDim.x + threadIdx.x;
  if (i < n) out[i] = b[i % C];
}

// ---------------------- weight pre-pack for B fragments --------------------
// B-fragment layout (16-bit B, 32x16 KxN, wave32), per ISA 7.12.2:
//   lane L: column n = L&15, element s -> k = (L>>4)*8 + (s&7) + (s>>3)*16.
// Pack so each lane reads its 16 bf16 as two contiguous b128 loads.
template<int KDIM, int NCOLS>
__global__ void k_pack_w(const float* __restrict__ W, bf16* __restrict__ Wp) {
  constexpr int NT = NCOLS / 16;
  constexpr int KC = KDIM / 32;
  unsigned i = blockIdx.x * blockDim.x + threadIdx.x;
  if (i >= (unsigned)(KC * NT * 512)) return;
  unsigned idx  = i & 15u;
  unsigned lane = (i >> 4) & 31u;
  unsigned tile = i >> 9;
  unsigned nt = tile % NT, kc = tile / NT;
  unsigned n = nt * 16 + (lane & 15u);
  unsigned k = kc * 32 + ((lane >> 4) * 8) + (idx & 7u) + ((idx >> 3) * 16);
  Wp[i] = f2bf(W[(size_t)k * NCOLS + n]);
}

// ------------------------------- WMMA GEMM ---------------------------------
// C[M,NCOLS] = A[M,KDIM](bf16,row-major) x Wp(packed bf16), f32 accumulate.
// One wave per 16x16 C tile; block = NT waves covering all column tiles of a
// 16-row stripe. A fragment (16x32): lane L holds row m=L&15; VGPRs 0..3 are
// k = half*8..+7 and VGPRs 4..7 are k = 16+half*8..+7  -> two b128 loads.
template<int KDIM, int NCOLS>
__global__ void k_wmma_gemm(const bf16* __restrict__ A, const bf16* __restrict__ Wp,
                            float* __restrict__ C) {
  constexpr int NT = NCOLS / 16;
  constexpr int KC = KDIM / 32;
  const int mt   = blockIdx.x;
  const int nt   = threadIdx.x >> 5;      // wave id == column tile
  const int lane = threadIdx.x & 31;
  const int half = lane >> 4;

  const bf16* ap = A  + (size_t)(mt * 16 + (lane & 15)) * KDIM + half * 8;
  const bf16* bp = Wp + (size_t)nt * 512 + (size_t)lane * 16;

  v8f acc = {};
#pragma unroll
  for (int kc = 0; kc < KC; ++kc) {
    Frag16 fa, fb;
    fa.u4[0] = *(const uint4*)(ap);
    fa.u4[1] = *(const uint4*)(ap + 16);
    fb.u4[0] = *(const uint4*)(bp);
    fb.u4[1] = *(const uint4*)(bp + 8);
    ap += 32;
    bp += (size_t)NT * 512;
    acc = __builtin_amdgcn_wmma_f32_16x16x32_bf16(
        false, fa.v, false, fb.v, (short)0, acc, false, false);
  }
  // C layout: lanes 0-15 -> M = j, lanes 16-31 -> M = 8+j; N = lane&15
  const int n  = nt * 16 + (lane & 15);
  const int mb = mt * 16 + half * 8;
#pragma unroll
  for (int j = 0; j < 8; ++j)
    C[(size_t)(mb + j) * NCOLS + n] = acc[j];
}

// --------------------------- attention scores ------------------------------
template<int H, int C>
__global__ void k_scores(const float* __restrict__ Hm, const float* __restrict__ a_src,
                         const float* __restrict__ a_dst, float* __restrict__ s_src,
                         float* __restrict__ s_dst, int nNodes) {
  int i = blockIdx.x * blockDim.x + threadIdx.x;
  if (i >= nNodes * H) return;
  int n = i / H, h = i % H;
  const float* row = Hm + (size_t)n * (H * C) + (size_t)h * C;
  float ss = 0.f, sd = 0.f;
#pragma unroll 8
  for (int c = 0; c < C; ++c) {
    float v = row[c];
    ss += v * a_src[h * C + c];
    sd += v * a_dst[h * C + c];
  }
  s_src[i] = ss;
  s_dst[i] = sd;
}

// ------------------------- edge passes (segment softmax) -------------------
template<int H>
__global__ void k_edge_max(const long long* __restrict__ ei, int E, int nNodes,
                           const float* __restrict__ s_src, const float* __restrict__ s_dst,
                           float* __restrict__ ew, unsigned* __restrict__ menc) {
  int e = blockIdx.x * blockDim.x + threadIdx.x;
  if (e >= E + nNodes) return;
  long long s, d;
  if (e < E) { s = ei[e]; d = ei[E + e]; } else { s = d = e - E; }
#pragma unroll
  for (int h = 0; h < H; ++h) {
    float v = s_src[s * H + h] + s_dst[d * H + h];
    v = (v > 0.f) ? v : 0.2f * v;            // LeakyReLU(0.2)
    ew[(size_t)e * H + h] = v;
    atomicMax(&menc[d * H + h], fenc(v));
  }
}

template<int H>
__global__ void k_edge_expsum(const long long* __restrict__ ei, int E, int nNodes,
                              float* __restrict__ ew, const unsigned* __restrict__ menc,
                              float* __restrict__ den) {
  int e = blockIdx.x * blockDim.x + threadIdx.x;
  if (e >= E + nNodes) return;
  long long d;
  if (e < E) { d = ei[E + e]; } else { d = e - E; }
#pragma unroll
  for (int h = 0; h < H; ++h) {
    float v = expf(ew[(size_t)e * H + h] - fdec(menc[d * H + h]));
    ew[(size_t)e * H + h] = v;
    atomicAdd(&den[d * H + h], v);
  }
}

// out[dst] += alpha * H[src]; one thread per (edge, head, 4-channel group)
template<int H, int C>
__global__ void k_agg(const long long* __restrict__ ei, int E, int nNodes,
                      const float* __restrict__ ew, const float* __restrict__ den,
                      const float* __restrict__ Hm, float* __restrict__ out) {
  constexpr int PE = H * C / 4;
  unsigned i = blockIdx.x * blockDim.x + threadIdx.x;
  unsigned EP = (unsigned)(E + nNodes);
  if (i >= EP * (unsigned)PE) return;
  unsigned e = i / PE, r = i % PE;
  int h = r / (C / 4), c4 = (r % (C / 4)) * 4;
  long long s, d;
  if (e < (unsigned)E) { s = ei[e]; d = ei[E + e]; } else { s = d = (long long)e - E; }
  float alpha = ew[(size_t)e * H + h] / den[d * H + h];
  const float4 hv = *(const float4*)(Hm + ((size_t)s * H + h) * C + c4);
  float* o = out + ((size_t)d * H + h) * C + c4;
  atomicAdd(o + 0, hv.x * alpha);
  atomicAdd(o + 1, hv.y * alpha);
  atomicAdd(o + 2, hv.z * alpha);
  atomicAdd(o + 3, hv.w * alpha);
}

// ------------------------- log_softmax, 64 cols, 1 wave/node ---------------
__global__ void k_logsoftmax64(float* __restrict__ out, int nNodes) {
  int w = (int)((blockIdx.x * blockDim.x + threadIdx.x) >> 5);
  int lane = threadIdx.x & 31;
  if (w >= nNodes) return;
  float* row = out + (size_t)w * 64;
  float v0 = row[lane], v1 = row[lane + 32];
  float m = fmaxf(v0, v1);
  for (int o = 16; o > 0; o >>= 1) m = fmaxf(m, __shfl_xor(m, o, 32));
  float s = expf(v0 - m) + expf(v1 - m);
  for (int o = 16; o > 0; o >>= 1) s += __shfl_xor(s, o, 32);
  float lse = m + logf(s);
  row[lane]      = v0 - lse;
  row[lane + 32] = v1 - lse;
}

// ---------------------------------------------------------------------------

extern "C" void kernel_launch(void* const* d_in, const int* in_sizes, int n_in,
                              void* d_out, int out_size, void* d_ws, size_t ws_size,
                              hipStream_t stream) {
  const float*     x      = (const float*)d_in[0];
  const long long* ei     = (const long long*)d_in[1];   // int64 [2,E]
  const float*     W1     = (const float*)d_in[2];
  const float*     a_src1 = (const float*)d_in[3];
  const float*     a_dst1 = (const float*)d_in[4];
  const float*     b1     = (const float*)d_in[5];
  const float*     W2     = (const float*)d_in[6];
  const float*     a_src2 = (const float*)d_in[7];
  const float*     a_dst2 = (const float*)d_in[8];
  const float*     b2     = (const float*)d_in[9];
  float* out = (float*)d_out;

  const int N  = in_sizes[0] / 128;   // 50000 (multiple of 16)
  const int E  = in_sizes[1] / 2;     // 800000
  const int EP = E + N;               // with self-loops

  // ---- workspace bump allocation (with liveness-based reuse), ~137 MB ----
  unsigned char* base = (unsigned char*)d_ws;
  size_t off = 0;
  auto alloc = [&](size_t bytes) -> void* {
    void* p = base + off;
    off += (bytes + 255) & ~(size_t)255;
    return p;
  };
  bf16*     w1p   = (bf16*)alloc((size_t)4 * 16 * 512 * 2);   // packed W1
  bf16*     w2p   = (bf16*)alloc((size_t)8 * 4 * 512 * 2);    // packed W2
  float*    ssrc1 = (float*)alloc((size_t)N * 4 * 4);
  float*    sdst1 = (float*)alloc((size_t)N * 4 * 4);
  unsigned* m1    = (unsigned*)alloc((size_t)N * 4 * 4);
  float*    den1  = (float*)alloc((size_t)N * 4 * 4);
  float*    ssrc2 = (float*)alloc((size_t)N * 4);
  float*    sdst2 = (float*)alloc((size_t)N * 4);
  unsigned* m2    = (unsigned*)alloc((size_t)N * 4);
  float*    den2  = (float*)alloc((size_t)N * 4);
  float*    ew1   = (float*)alloc((size_t)EP * 4 * 4);  // reused later as H2
  float*    ew2   = (float*)alloc((size_t)EP * 4);
  bf16*     xb    = (bf16*)alloc((size_t)N * 128 * 2);  // reused later as hb
  float*    H1    = (float*)alloc((size_t)N * 256 * 4); // adjacent to xb
  float*    out1  = (float*)alloc((size_t)N * 256 * 4);
  // aliases (dead-region reuse):
  bf16*  hb = xb;          // 25.6MB over dead xb(12.8) + head of dead H1
  float* H2 = ew1;         // 12.8MB over dead ew1(13.6)

  const unsigned ENC_NEG_INF = 0x007FFFFFu;  // fenc(-inf)
  const int B = 256;
  auto blk = [](unsigned n, int b) { return (unsigned)((n + b - 1) / b); };

  // ---------------- layer 1 ----------------
  k_f32_to_bf16<<<blk((unsigned)N * 128, B), B, 0, stream>>>(x, xb, (unsigned)N * 128);
  k_pack_w<128, 256><<<blk(4 * 16 * 512, B), B, 0, stream>>>(W1, w1p);
  k_pack_w<256, 64><<<blk(8 * 4 * 512, B), B, 0, stream>>>(W2, w2p);

  k_wmma_gemm<128, 256><<<dim3(N / 16), dim3(512), 0, stream>>>(xb, w1p, H1);

  k_scores<4, 64><<<blk((unsigned)N * 4, B), B, 0, stream>>>(H1, a_src1, a_dst1,
                                                             ssrc1, sdst1, N);
  k_fill_u32<<<blk((unsigned)N * 4, B), B, 0, stream>>>(m1, ENC_NEG_INF, (unsigned)N * 4);
  k_fill_u32<<<blk((unsigned)N * 4, B), B, 0, stream>>>((unsigned*)den1, 0u, (unsigned)N * 4);
  k_bias_init<<<blk((unsigned)N * 256, B), B, 0, stream>>>(out1, b1, (unsigned)N * 256, 256);

  k_edge_max<4><<<blk((unsigned)EP, B), B, 0, stream>>>(ei, E, N, ssrc1, sdst1, ew1, m1);
  k_edge_expsum<4><<<blk((unsigned)EP, B), B, 0, stream>>>(ei, E, N, ew1, m1, den1);
  k_agg<4, 64><<<blk((unsigned)EP * 64u, B), B, 0, stream>>>(ei, E, N, ew1, den1, H1, out1);

  k_elu_to_bf16<<<blk((unsigned)N * 256, B), B, 0, stream>>>(out1, hb, (unsigned)N * 256);

  // ---------------- layer 2 ----------------
  k_wmma_gemm<256, 64><<<dim3(N / 16), dim3(128), 0, stream>>>(hb, w2p, H2);

  k_scores<1, 64><<<blk((unsigned)N, B), B, 0, stream>>>(H2, a_src2, a_dst2,
                                                         ssrc2, sdst2, N);
  k_fill_u32<<<blk((unsigned)N, B), B, 0, stream>>>(m2, ENC_NEG_INF, (unsigned)N);
  k_fill_u32<<<blk((unsigned)N, B), B, 0, stream>>>((unsigned*)den2, 0u, (unsigned)N);
  k_bias_init<<<blk((unsigned)N * 64, B), B, 0, stream>>>(out, b2, (unsigned)N * 64, 64);

  k_edge_max<1><<<blk((unsigned)EP, B), B, 0, stream>>>(ei, E, N, ssrc2, sdst2, ew2, m2);
  k_edge_expsum<1><<<blk((unsigned)EP, B), B, 0, stream>>>(ei, E, N, ew2, m2, den2);
  k_agg<1, 64><<<blk((unsigned)EP * 16u, B), B, 0, stream>>>(ei, E, N, ew2, den2, H2, out);

  k_logsoftmax64<<<blk((unsigned)N * 32u, B), B, 0, stream>>>(out, N);
}
